// Model_86242943303881
// MI455X (gfx1250) — compile-verified, matrix operand order checked by reading, not run on hardware
//
#include <hip/hip_runtime.h>
#include <cstdint>
#include <cstddef>

// ---------------- problem dims (fixed by reference) ----------------
#define NV    4096   // n_vertices  (M of every GEMM; multiple of BM)
#define EDIM  100    // embedding dim
#define HDIM  400    // lstm hidden per direction
#define GDIM  1600   // 4*HDIM gates
#define MHDIM 1024   // mlp hidden
#define DDIM  500    // feature dim

// ---------------- WMMA GEMM tiling ----------------
#define BM 128
#define BN 64
#define BK 32
#define LDSK 40      // padded LDS row stride in halves (80B pitch: 16B aligned, banks spread)

typedef _Float16 v16h __attribute__((ext_vector_type(16)));
typedef float    v8f  __attribute__((ext_vector_type(8)));

enum { BLAY_NT = 0, BLAY_NN = 1 };
enum { EP_NONE = 0, EP_BIAS2 = 1, EP_BIAS_RELU = 2, EP_LEAKY = 3 };

union FragU { v16h v; uint4 q[2]; };
union H4   { _Float16 h[4]; unsigned long long u; };

// Fragment fetch from an LDS tile laid out as [row][LDSK] halves.
// CDNA5 16-bit A/B layout: lanes 0-15 hold K {0..7,16..23}, lanes 16-31 hold
// K {8..15,24..31}; row = tile_row + lane%16.  -> two ds_load_b128 per frag.
__device__ __forceinline__ v16h frag_load(const _Float16* t, int row, int lane) {
  int lr = row + (lane & 15);
  int kb = (lane < 16) ? 0 : 8;
  const _Float16* p = t + lr * LDSK + kb;
  FragU f;
  f.q[0] = *(const uint4*)(p);
  f.q[1] = *(const uint4*)(p + 16);
  return f.v;
}

// split-f16 triple-WMMA: C += Ahi*Bhi + Ahi*Blo + Alo*Bhi  (~fp32 accuracy)
__device__ __forceinline__ v8f wmma3(v16h ah, v16h al, v16h bh, v16h bl, v8f c) {
  c = __builtin_amdgcn_wmma_f32_16x16x32_f16(false, ah, false, bh, (short)0, c, false, false);
  c = __builtin_amdgcn_wmma_f32_16x16x32_f16(false, ah, false, bl, (short)0, c, false, false);
  c = __builtin_amdgcn_wmma_f32_16x16x32_f16(false, al, false, bh, (short)0, c, false, false);
  return c;
}

// C[M,N] = A[M,K] * B (+epilogue), fp32 in/out, split-f16 WMMA inside.
// BLAY_NT: B row-major [N,K]  (C = A*B^T, x@W.T pattern)
// BLAY_NN: B row-major [K,N]  (C = A*B)
// Requires: M multiple of BM (all call sites use M = 4096).
template<int BLAY, int EP, bool ACC>
__global__ __launch_bounds__(256) void gemm_wmma(
    const float* __restrict__ A, int lda,
    const float* __restrict__ B, int ldb,
    float* __restrict__ C, int ldc,
    int M, int N, int K,
    const float* __restrict__ b1, const float* __restrict__ b2)
{
  __shared__ __align__(16) _Float16 sAh[2][BM * LDSK];
  __shared__ __align__(16) _Float16 sAl[2][BM * LDSK];
  __shared__ __align__(16) _Float16 sBh[2][BN * LDSK];
  __shared__ __align__(16) _Float16 sBl[2][BN * LDSK];

  const int t  = threadIdx.x;
  const int m0 = blockIdx.y * BM;
  const int n0 = blockIdx.x * BN;
  const int wave = t >> 5;
  const int lane = t & 31;
  const int wm = (wave & 3) * 32;   // 4 wave-rows of 32
  const int wn = (wave >> 2) * 32;  // 2 wave-cols of 32
  const bool n_full = (n0 + BN <= N);

  v8f zero = {0.f,0.f,0.f,0.f,0.f,0.f,0.f,0.f};
  v8f acc[2][2];
  acc[0][0] = zero; acc[0][1] = zero; acc[1][0] = zero; acc[1][1] = zero;

  const int ktiles = (K + BK - 1) / BK;

  float ra[16];   // staged A elements (register double-buffer)
  float rb[8];    // staged B elements

  // ---- issue global loads for k-tile kt into ra/rb (branchless edge path) ----
  auto load_tiles = [&](int kt) {
    const int k0 = kt * BK;
    const bool k_full = (k0 + BK <= K);
    // A: 128x32, mapping: idx4 = t+256*i -> row = idx4>>3, k4 = (idx4&7)*4
    if (k_full) {
      #pragma unroll
      for (int i = 0; i < 4; ++i) {
        int idx4 = t + 256 * i;
        int r = idx4 >> 3, k4 = (idx4 & 7) * 4;
        float4 v = *(const float4*)(A + (size_t)(m0 + r) * lda + k0 + k4);
        ra[4*i+0] = v.x; ra[4*i+1] = v.y; ra[4*i+2] = v.z; ra[4*i+3] = v.w;
      }
    } else {
      #pragma unroll
      for (int i = 0; i < 4; ++i) {
        int idx4 = t + 256 * i;
        int r = idx4 >> 3, k4 = (idx4 & 7) * 4;
        #pragma unroll
        for (int j = 0; j < 4; ++j) {
          int gk = k0 + k4 + j;
          int ck = (gk < K) ? gk : (K - 1);              // clamp: valid address
          float v = A[(size_t)(m0 + r) * lda + ck];
          ra[4*i+j] = (gk < K) ? v : 0.0f;               // v_cndmask, no branch
        }
      }
    }
    if (BLAY == BLAY_NT) {
      // B: 64 rows (n) x 32 (k); idx4 -> n = idx4>>3, k4 = (idx4&7)*4
      if (k_full && n_full) {
        #pragma unroll
        for (int i = 0; i < 2; ++i) {
          int idx4 = t + 256 * i;
          int n = idx4 >> 3, k4 = (idx4 & 7) * 4;
          float4 v = *(const float4*)(B + (size_t)(n0 + n) * ldb + k0 + k4);
          rb[4*i+0] = v.x; rb[4*i+1] = v.y; rb[4*i+2] = v.z; rb[4*i+3] = v.w;
        }
      } else {
        #pragma unroll
        for (int i = 0; i < 2; ++i) {
          int idx4 = t + 256 * i;
          int n = idx4 >> 3, k4 = (idx4 & 7) * 4;
          int gn = n0 + n;
          int cn = (gn < N) ? gn : (N - 1);
          #pragma unroll
          for (int j = 0; j < 4; ++j) {
            int gk = k0 + k4 + j;
            int ck = (gk < K) ? gk : (K - 1);
            float v = B[(size_t)cn * ldb + ck];
            rb[4*i+j] = (gn < N && gk < K) ? v : 0.0f;
          }
        }
      }
    } else {
      // B: [K,N]; float4 along n: idx4 -> k = idx4>>4, n4 = (idx4&15)*4
      if (k_full && n_full) {
        #pragma unroll
        for (int i = 0; i < 2; ++i) {
          int idx4 = t + 256 * i;
          int k = idx4 >> 4, n4 = (idx4 & 15) * 4;
          float4 v = *(const float4*)(B + (size_t)(k0 + k) * ldb + n0 + n4);
          rb[4*i+0] = v.x; rb[4*i+1] = v.y; rb[4*i+2] = v.z; rb[4*i+3] = v.w;
        }
      } else {
        #pragma unroll
        for (int i = 0; i < 2; ++i) {
          int idx4 = t + 256 * i;
          int k = idx4 >> 4, n4 = (idx4 & 15) * 4;
          int gk = k0 + k;
          int ck = (gk < K) ? gk : (K - 1);
          #pragma unroll
          for (int j = 0; j < 4; ++j) {
            int gn = n0 + n4 + j;
            int cn = (gn < N) ? gn : (N - 1);
            float v = B[(size_t)ck * ldb + cn];
            rb[4*i+j] = (gk < K && gn < N) ? v : 0.0f;
          }
        }
      }
    }
  };

  // ---- convert fp32 -> (hi,lo) f16 and store to LDS stage s ----
  auto store_tiles = [&](int s) {
    #pragma unroll
    for (int i = 0; i < 4; ++i) {
      int idx4 = t + 256 * i;
      int r = idx4 >> 3, k4 = (idx4 & 7) * 4;
      H4 hh, hl;
      #pragma unroll
      for (int j = 0; j < 4; ++j) {
        float v = ra[4*i+j];
        _Float16 hi = (_Float16)v;
        hh.h[j] = hi;
        hl.h[j] = (_Float16)(v - (float)hi);
      }
      *(unsigned long long*)&sAh[s][r * LDSK + k4] = hh.u;   // ds_store_b64
      *(unsigned long long*)&sAl[s][r * LDSK + k4] = hl.u;
    }
    if (BLAY == BLAY_NT) {
      #pragma unroll
      for (int i = 0; i < 2; ++i) {
        int idx4 = t + 256 * i;
        int n = idx4 >> 3, k4 = (idx4 & 7) * 4;
        H4 hh, hl;
        #pragma unroll
        for (int j = 0; j < 4; ++j) {
          float v = rb[4*i+j];
          _Float16 hi = (_Float16)v;
          hh.h[j] = hi;
          hl.h[j] = (_Float16)(v - (float)hi);
        }
        *(unsigned long long*)&sBh[s][n * LDSK + k4] = hh.u;
        *(unsigned long long*)&sBl[s][n * LDSK + k4] = hl.u;
      }
    } else {
      #pragma unroll
      for (int i = 0; i < 2; ++i) {
        int idx4 = t + 256 * i;
        int k = idx4 >> 4, n4 = (idx4 & 15) * 4;
        #pragma unroll
        for (int j = 0; j < 4; ++j) {          // transposed store, scalar b16
          float v = rb[4*i+j];
          _Float16 hi = (_Float16)v;
          sBh[s][(n4 + j) * LDSK + k] = hi;
          sBl[s][(n4 + j) * LDSK + k] = (_Float16)(v - (float)hi);
        }
      }
    }
  };

  auto compute = [&](int s) {
    v16h ah0 = frag_load(sAh[s], wm,      lane);
    v16h ah1 = frag_load(sAh[s], wm + 16, lane);
    v16h al0 = frag_load(sAl[s], wm,      lane);
    v16h al1 = frag_load(sAl[s], wm + 16, lane);
    v16h bh0 = frag_load(sBh[s], wn,      lane);
    v16h bh1 = frag_load(sBh[s], wn + 16, lane);
    v16h bl0 = frag_load(sBl[s], wn,      lane);
    v16h bl1 = frag_load(sBl[s], wn + 16, lane);
    acc[0][0] = wmma3(ah0, al0, bh0, bl0, acc[0][0]);
    acc[0][1] = wmma3(ah0, al0, bh1, bl1, acc[0][1]);
    acc[1][0] = wmma3(ah1, al1, bh0, bl0, acc[1][0]);
    acc[1][1] = wmma3(ah1, al1, bh1, bl1, acc[1][1]);
  };

  // ---- double-buffered pipeline: overlap next-tile global loads with WMMA ----
  load_tiles(0);
  store_tiles(0);
  __syncthreads();
  for (int kt = 0; kt < ktiles; ++kt) {
    const int cur = kt & 1;
    const bool more = (kt + 1 < ktiles);
    if (more) load_tiles(kt + 1);   // global loads in flight during compute
    compute(cur);
    if (more) store_tiles(cur ^ 1);
    __syncthreads();
  }

  // ---- epilogue + store (elem r of v8f -> M = r + 8*(lane>>4), N = lane&15)
  // M is a multiple of BM at every call site -> only the N edge needs a check.
  const int mbase = m0 + wm + 8 * (lane >> 4);
  const int nbase = n0 + wn + (lane & 15);
  #pragma unroll
  for (int j = 0; j < 2; ++j)
    #pragma unroll
    for (int jj = 0; jj < 2; ++jj) {
      int gn = nbase + jj * 16;
      if (gn < N) {
        #pragma unroll
        for (int r = 0; r < 8; ++r) {
          int gm = mbase + j * 16 + r;
          float v = acc[j][jj][r];
          float* cp = &C[(size_t)gm * ldc + gn];
          if (ACC) v += *cp;
          if (EP == EP_BIAS2)     v += b1[gn] + b2[gn];
          if (EP == EP_BIAS_RELU) { v += b1[gn]; v = fmaxf(v, 0.0f); }
          if (EP == EP_LEAKY)     v = (v > 0.0f) ? v : 0.1f * v;
          *cp = v;
        }
      }
    }
}

// ---------------- small glue kernels ----------------
__device__ __forceinline__ float sigmoidf_(float x) { return 1.0f / (1.0f + __expf(-x)); }

__global__ __launch_bounds__(256) void embed_kernel(
    const int* __restrict__ word, const int* __restrict__ tag,
    const float* __restrict__ wemb, const float* __restrict__ temb,
    float* __restrict__ x0)
{
  int tid = blockIdx.x * 256 + threadIdx.x;
  if (tid >= NV * EDIM) return;
  int n = tid / EDIM, e = tid % EDIM;
  x0[(size_t)n * (2 * EDIM) + e]        = wemb[(size_t)word[n] * EDIM + e];
  x0[(size_t)n * (2 * EDIM) + EDIM + e] = temb[(size_t)tag[n] * EDIM + e];
}

// S=1 LSTM cell: c = sig(i)*tanh(g), h = sig(o)*tanh(c); write into [NV,800] slice
__global__ __launch_bounds__(256) void lstm_act_kernel(
    const float* __restrict__ g, float* __restrict__ out, int col_off)
{
  int tid = blockIdx.x * 256 + threadIdx.x;
  if (tid >= NV * HDIM) return;
  int n = tid / HDIM, j = tid % HDIM;
  const float* gr = g + (size_t)n * GDIM;
  float ig = gr[j], gg = gr[2 * HDIM + j], og = gr[3 * HDIM + j];
  float c = sigmoidf_(ig) * tanhf(gg);
  float h = sigmoidf_(og) * tanhf(c);
  out[(size_t)n * (2 * HDIM) + col_off + j] = h;
}

// r[row] = dot(X[row,:], v)   (one wave per row)
__global__ __launch_bounds__(256) void gemv_kernel(
    const float* __restrict__ X, const float* __restrict__ v,
    float* __restrict__ out, int nrows, int ncols)
{
  int w = threadIdx.x >> 5, lane = threadIdx.x & 31;
  int row = blockIdx.x * 8 + w;
  if (row >= nrows) return;
  float s = 0.0f;
  for (int j = lane; j < ncols; j += 32) s += X[(size_t)row * ncols + j] * v[j];
  #pragma unroll
  for (int o = 16; o; o >>= 1) s += __shfl_down(s, o);
  if (lane == 0) out[row] = s;
}

// In-place row softmax of s[i,j] + r2[j] over j (NV columns, held in regs)
__global__ __launch_bounds__(256) void softmax_kernel(
    float* __restrict__ s, const float* __restrict__ r2)
{
  __shared__ float sb[8];
  float* p = s + (size_t)blockIdx.x * NV;
  int lane = threadIdx.x & 31, w = threadIdx.x >> 5;
  float v[16];
  float mx = -3.4e38f;
  #pragma unroll
  for (int i = 0; i < 16; ++i) {
    int j = threadIdx.x + 256 * i;
    v[i] = p[j] + r2[j];
    mx = fmaxf(mx, v[i]);
  }
  #pragma unroll
  for (int o = 16; o; o >>= 1) mx = fmaxf(mx, __shfl_down(mx, o));
  if (lane == 0) sb[w] = mx;
  __syncthreads();
  mx = sb[0];
  #pragma unroll
  for (int i = 1; i < 8; ++i) mx = fmaxf(mx, sb[i]);
  __syncthreads();
  float sum = 0.0f;
  #pragma unroll
  for (int i = 0; i < 16; ++i) { v[i] = __expf(v[i] - mx); sum += v[i]; }
  #pragma unroll
  for (int o = 16; o; o >>= 1) sum += __shfl_down(sum, o);
  if (lane == 0) sb[w] = sum;
  __syncthreads();
  sum = 0.0f;
  #pragma unroll
  for (int i = 0; i < 8; ++i) sum += sb[i];
  float inv = 1.0f / sum;
  #pragma unroll
  for (int i = 0; i < 16; ++i) { int j = threadIdx.x + 256 * i; p[j] = v[i] * inv; }
}

__global__ void transpose_kernel(const float* __restrict__ in, float* __restrict__ out)
{
  __shared__ float tile[32][33];
  int bx = blockIdx.x * 32, by = blockIdx.y * 32;
  int tx = threadIdx.x, ty = threadIdx.y;   // block (32,8)
  #pragma unroll
  for (int i = 0; i < 32; i += 8)
    tile[ty + i][tx] = in[(size_t)(by + ty + i) * NV + bx + tx];
  __syncthreads();
  #pragma unroll
  for (int i = 0; i < 32; i += 8)
    out[(size_t)(bx + ty + i) * NV + by + tx] = tile[tx][ty + i];
}

__global__ __launch_bounds__(256) void diag_kernel(const float* __restrict__ a, float* __restrict__ dg)
{
  int i = blockIdx.x * 256 + threadIdx.x;
  if (i < NV) dg[i] = a[(size_t)i * NV + i];
}

// agg[i,j] -= diag(alpha)[i] * (head[i,j] + dep[i,j]) for both aggs
__global__ __launch_bounds__(256) void fixup_kernel(
    float* __restrict__ hagg, float* __restrict__ dagg,
    const float* __restrict__ head, const float* __restrict__ dep,
    const float* __restrict__ dg)
{
  int tid = blockIdx.x * 256 + threadIdx.x;
  if (tid >= NV * DDIM) return;
  int n = tid / DDIM;
  float t = dg[n] * (head[tid] + dep[tid]);
  hagg[tid] -= t;
  dagg[tid] -= t;
}

// ---------------- host-side helpers ----------------
template<int BLAY, int EP, bool ACC>
static void launch_gemm(hipStream_t st,
                        const float* A, int lda, const float* B, int ldb,
                        float* C, int ldc, int M, int N, int K,
                        const float* b1, const float* b2)
{
  dim3 grid((N + BN - 1) / BN, (M + BM - 1) / BM);
  gemm_wmma<BLAY, EP, ACC><<<grid, dim3(256), 0, st>>>(A, lda, B, ldb, C, ldc, M, N, K, b1, b2);
}

static void run_gnn(hipStream_t st, const float* alpha, float* alphaT,
                    const float* head, const float* dep,
                    const float* aggh, const float* aggd,
                    const float* combh, const float* combd,
                    float* hagg, float* dagg, float* dg,
                    float* hout, float* dout)
{
  transpose_kernel<<<dim3(NV / 32, NV / 32), dim3(32, 8), 0, st>>>(alpha, alphaT);
  // head_agg = alpha^T @ head + alpha @ dep
  launch_gemm<BLAY_NN, EP_NONE, false>(st, alphaT, NV, head, DDIM, hagg, DDIM, NV, DDIM, NV, nullptr, nullptr);
  launch_gemm<BLAY_NN, EP_NONE, true >(st, alpha,  NV, dep,  DDIM, hagg, DDIM, NV, DDIM, NV, nullptr, nullptr);
  // dep_agg = alpha @ head + alpha^T @ dep
  launch_gemm<BLAY_NN, EP_NONE, false>(st, alpha,  NV, head, DDIM, dagg, DDIM, NV, DDIM, NV, nullptr, nullptr);
  launch_gemm<BLAY_NN, EP_NONE, true >(st, alphaT, NV, dep,  DDIM, dagg, DDIM, NV, DDIM, NV, nullptr, nullptr);
  diag_kernel<<<NV / 256, 256, 0, st>>>(alpha, dg);
  fixup_kernel<<<(NV * DDIM + 255) / 256, 256, 0, st>>>(hagg, dagg, head, dep, dg);
  // h = leaky_relu(head_agg @ aw_h + head @ cw_h, 0.1)
  launch_gemm<BLAY_NN, EP_NONE,  false>(st, hagg, DDIM, aggh,  DDIM, hout, DDIM, NV, DDIM, DDIM, nullptr, nullptr);
  launch_gemm<BLAY_NN, EP_LEAKY, true >(st, head, DDIM, combh, DDIM, hout, DDIM, NV, DDIM, DDIM, nullptr, nullptr);
  launch_gemm<BLAY_NN, EP_NONE,  false>(st, dagg, DDIM, aggd,  DDIM, dout, DDIM, NV, DDIM, DDIM, nullptr, nullptr);
  launch_gemm<BLAY_NN, EP_LEAKY, true >(st, dep,  DDIM, combd, DDIM, dout, DDIM, NV, DDIM, DDIM, nullptr, nullptr);
}

extern "C" void kernel_launch(void* const* d_in, const int* in_sizes, int n_in,
                              void* d_out, int out_size, void* d_ws, size_t ws_size,
                              hipStream_t stream)
{
  (void)in_sizes; (void)n_in; (void)out_size; (void)ws_size;
  auto F = [&](int i) { return (const float*)d_in[i]; };
  const int* word = (const int*)d_in[0];
  const int* tag  = (const int*)d_in[1];

  // ---- workspace plan (floats). Region1 (lstm/mlp) is reused later as alphaT.
  float* ws    = (float*)d_ws;
  float* x0    = ws;                       // 819200
  float* xA    = ws + 819200;              // 3276800
  float* xB    = xA + 3276800;             // 3276800
  float* gates = xB + 3276800;             // 6553600
  float* hid   = gates + 6553600;          // 4194304  -> region1 end = 18120704
  float* alphaT = ws;                      // alias region1 (16777216 <= 18120704)
  float* headA = ws + 18120704;
  float* depA  = headA + (size_t)NV * DDIM;
  float* headB = depA  + (size_t)NV * DDIM;
  float* depB  = headB + (size_t)NV * DDIM;
  float* Tbuf  = depB  + (size_t)NV * DDIM;  // scores temp / hagg
  float* dagg  = Tbuf  + (size_t)NV * DDIM;
  float* r2    = dagg  + (size_t)NV * DDIM;
  float* dg    = r2 + NV;

  float* out_head = (float*)d_out;
  float* out_dep  = out_head + (size_t)NV * DDIM;
  float* alpha1   = out_dep  + (size_t)NV * DDIM;
  float* alpha2   = alpha1   + (size_t)NV * NV;

  // 1) embeddings -> x0 [NV, 200]
  embed_kernel<<<(NV * EDIM + 255) / 256, 256, 0, stream>>>(word, tag, F(2), F(3), x0);

  // 2) 3 BiLSTM layers, S=1 => gates GEMM + elementwise cell per direction
  const float* xin = x0; int lda = 2 * EDIM; int Kin = 2 * EDIM;
  float* xout = xA;
  for (int l = 0; l < 3; ++l) {
    const float* Wih = F(4 + 3 * l);
    const float* bb  = F(6 + 3 * l);
    for (int dir = 0; dir < 2; ++dir) {
      const float* W  = Wih + (size_t)dir * GDIM * Kin;
      const float* b1 = bb + (size_t)dir * 2 * GDIM;   // bih
      const float* b2 = b1 + GDIM;                     // bhh
      launch_gemm<BLAY_NT, EP_BIAS2, false>(stream, xin, lda, W, Kin, gates, GDIM,
                                            NV, GDIM, Kin, b1, b2);
      lstm_act_kernel<<<(NV * HDIM + 255) / 256, 256, 0, stream>>>(gates, xout, dir * HDIM);
    }
    xin = xout; lda = 2 * HDIM; Kin = 2 * HDIM;
    xout = (xout == xA) ? xB : xA;
  }
  const float* xfin = xin;                 // = xA after 3 layers
  const float* head0 = xfin;               // cols [0,400), lda 800
  const float* dep0  = xfin + HDIM;        // cols [400,800), lda 800

  // 3) ArcMLPs: relu(relu(x@W1.T+b1)@W2.T+b2)
  launch_gemm<BLAY_NT, EP_BIAS_RELU, false>(stream, head0, 2 * HDIM, F(13), HDIM, hid, MHDIM,
                                            NV, MHDIM, HDIM, F(14), nullptr);
  launch_gemm<BLAY_NT, EP_BIAS_RELU, false>(stream, hid, MHDIM, F(15), MHDIM, headA, DDIM,
                                            NV, DDIM, MHDIM, F(16), nullptr);
  launch_gemm<BLAY_NT, EP_BIAS_RELU, false>(stream, dep0, 2 * HDIM, F(17), HDIM, hid, MHDIM,
                                            NV, MHDIM, HDIM, F(18), nullptr);
  launch_gemm<BLAY_NT, EP_BIAS_RELU, false>(stream, hid, MHDIM, F(19), MHDIM, depA, DDIM,
                                            NV, DDIM, MHDIM, F(20), nullptr);

  // 4) transprob1: softmax_j(head@A@dep.T + dep@b2[j])  (head@b1 term cancels in softmax)
  launch_gemm<BLAY_NN, EP_NONE, false>(stream, headA, DDIM, F(21), DDIM, Tbuf, DDIM,
                                       NV, DDIM, DDIM, nullptr, nullptr);
  gemv_kernel<<<NV / 8, 256, 0, stream>>>(depA, F(23), r2, NV, DDIM);
  launch_gemm<BLAY_NT, EP_NONE, false>(stream, Tbuf, DDIM, depA, DDIM, alpha1, NV,
                                       NV, NV, DDIM, nullptr, nullptr);
  softmax_kernel<<<NV, 256, 0, stream>>>(alpha1, r2);

  // 5) GNN layer 1 -> headB/depB
  run_gnn(stream, alpha1, alphaT, headA, depA, F(27), F(28), F(29), F(30),
          Tbuf, dagg, dg, headB, depB);

  // 6) transprob2
  launch_gemm<BLAY_NN, EP_NONE, false>(stream, headB, DDIM, F(24), DDIM, Tbuf, DDIM,
                                       NV, DDIM, DDIM, nullptr, nullptr);
  gemv_kernel<<<NV / 8, 256, 0, stream>>>(depB, F(26), r2, NV, DDIM);
  launch_gemm<BLAY_NT, EP_NONE, false>(stream, Tbuf, DDIM, depB, DDIM, alpha2, NV,
                                       NV, NV, DDIM, nullptr, nullptr);
  softmax_kernel<<<NV, 256, 0, stream>>>(alpha2, r2);

  // 7) GNN layer 2 -> final head/dep straight into d_out
  run_gnn(stream, alpha2, alphaT, headB, depB, F(31), F(32), F(33), F(34),
          Tbuf, dagg, dg, out_head, out_dep);
}
